// GeographicDailyCrossSourceFusion_18957985645085
// MI455X (gfx1250) — matrix operand order repaired and checked by reading, not compiled, not run on hardware
//
#include <hip/hip_runtime.h>
#include <hip/hip_bf16.h>
#include <math.h>

// ---------------------------------------------------------------------------
// CDNA5 WMMA helpers (gfx1250, wave32)
// D[16,16](f32) = A[16,32](f16) x B[32,16](f16) + C
// A frag: lane&15 = m, (lane>>4)*8 = k-base, elem j -> k = kb + j + (j>=8 ? 8 : 0)
// B frag: stored as row-major [K][N] (i.e. W^T), lane&15 = n, elem j -> k as in A frag
// C/D   : elem i -> m = (lane>>4)*8 + i, n = lane&15
// ---------------------------------------------------------------------------

typedef __attribute__((ext_vector_type(16))) _Float16 v16h;
typedef __attribute__((ext_vector_type(8)))  _Float16 v8h;
typedef __attribute__((ext_vector_type(8)))  float    v8f;

#define DEV static __device__ __forceinline__

DEV v8f wmma_f16(v16h a, v16h b, v8f c) {
  return __builtin_amdgcn_wmma_f32_16x16x32_f16(false, a, false, b, (short)0, c,
                                                false, false);
}

DEV v16h loadA_h(const _Float16* A, int lda, int mBase, int kBase) {
  const int l = threadIdx.x & 31;
  const _Float16* p = A + (size_t)(mBase + (l & 15)) * lda + kBase + ((l >> 4) << 3);
  v8h lo = *(const v8h*)p;
  v8h hi = *(const v8h*)(p + 16);
  return __builtin_shufflevector(lo, hi, 0, 1, 2, 3, 4, 5, 6, 7,
                                 8, 9, 10, 11, 12, 13, 14, 15);
}

DEV v16h loadA_f(const float* A, int lda, int mBase, int kBase) {
  const int l = threadIdx.x & 31;
  const float* p = A + (size_t)(mBase + (l & 15)) * lda + kBase + ((l >> 4) << 3);
  v8f lo = *(const v8f*)p;
  v8f hi = *(const v8f*)(p + 16);
  v8h l2 = __builtin_convertvector(lo, v8h);
  v8h h2 = __builtin_convertvector(hi, v8h);
  return __builtin_shufflevector(l2, h2, 0, 1, 2, 3, 4, 5, 6, 7,
                                 8, 9, 10, 11, 12, 13, 14, 15);
}

DEV v16h loadB_h(const _Float16* Bt, int ldb, int kBase, int nBase) {
  const int l = threadIdx.x & 31;
  v16h r;
#pragma unroll
  for (int s = 0; s < 16; ++s)
    r[s] = Bt[(size_t)(kBase + ((s >> 3) << 4) + ((l >> 4) << 3) + (s & 7)) * ldb + nBase + (l & 15)];
  return r;
}

DEV float gelu_f(float x) { return 0.5f * x * (1.f + erff(x * 0.7071067811865475f)); }

DEV float wred_sum(float v) {
#pragma unroll
  for (int d = 1; d < 32; d <<= 1) v += __shfl_xor(v, d, 32);
  return v;
}
DEV float hred_sum(float v) {  // reduce across the 16 lanes of a half-wave
#pragma unroll
  for (int d = 1; d < 16; d <<= 1) v += __shfl_xor(v, d, 32);
  return v;
}
DEV float hred_max(float v) {
#pragma unroll
  for (int d = 1; d < 16; d <<= 1) v = fmaxf(v, __shfl_xor(v, d, 32));
  return v;
}

DEV void wave_ln_lds(float* row, int width, const float* g, const float* b, bool do_gelu) {
  const int l = threadIdx.x & 31;
  float s = 0.f;
  for (int c = l; c < width; c += 32) s += row[c];
  s = wred_sum(s);
  const float mu = s / (float)width;
  float var = 0.f;
  for (int c = l; c < width; c += 32) { float d = row[c] - mu; var += d * d; }
  var = wred_sum(var) / (float)width;
  const float rstd = rsqrtf(var + 1e-5f);
  for (int c = l; c < width; c += 32) {
    float y = g[c] * (row[c] - mu) * rstd + b[c];
    row[c] = do_gelu ? gelu_f(y) : y;
  }
}

// ---------------------------------------------------------------------------
// K0: weight convert + transpose: W[out][in] f32 -> Wt[in][out] f16
// ---------------------------------------------------------------------------
__global__ void __launch_bounds__(256) cvt_transpose(const float* __restrict__ W,
                                                     _Float16* __restrict__ Wt,
                                                     int rows, int cols) {
  int idx = blockIdx.x * 256 + threadIdx.x;
  if (idx < rows * cols) {
    int r = idx / cols, c = idx % cols;
    Wt[(size_t)c * rows + r] = (_Float16)W[(size_t)r * cols + c];
  }
}

// ---------------------------------------------------------------------------
// K1: region projection + geographic cross attention + LN + region mean
// one workgroup per (b,s); R=136 padded to 144; D=128; H=8; hd=16
// ---------------------------------------------------------------------------
__global__ void __launch_bounds__(256) encode_regions(
    const float* __restrict__ xv, const float* __restrict__ geo,
    const float* __restrict__ priorw,
    const _Float16* __restrict__ wrp_t, const float* __restrict__ b_rp,
    const float* __restrict__ remb,
    const _Float16* __restrict__ win_t, const float* __restrict__ b_in,
    const _Float16* __restrict__ wo_t, const float* __restrict__ b_o,
    const float* __restrict__ ln1g, const float* __restrict__ ln1b,
    float* __restrict__ x_agg, _Float16* __restrict__ x_agg_h) {
  const int bs = blockIdx.x;
  const int tid = threadIdx.x, w = tid >> 5, l = tid & 31;
  extern __shared__ char smem[];
  float*    XP = (float*)smem;                  // [144][128] f32   73728 B
  _Float16* Qm = (_Float16*)(smem + 73728);     // [144][128] f16   36864 B
  _Float16* Kt = (_Float16*)(smem + 110592);    // [128][144] (transposed K)
  _Float16* Vm = (_Float16*)(smem + 147456);    // [144][128]
  _Float16* Pm = (_Float16*)(smem + 184320);    // [144][160]       46080 B
  _Float16* Om = (_Float16*)(smem + 230400);    // [144][128]       36864 B  (end 267264)
  float*    Xs = (float*)Pm;                    // [144][32] f32 staging (alias)

  // stage 0: async-copy the contiguous 136x32 f32 slice into LDS (ASYNCcnt path)
  {
    const float* src = xv + (size_t)bs * (136 * 32);
    for (int c4 = tid; c4 < 1088; c4 += 256) {  // 1088 x 16B chunks
      unsigned long long sp = (unsigned long long)(uintptr_t)(src + c4 * 4);
      unsigned dp = (unsigned)(uintptr_t)(Xs + c4 * 4);
      asm volatile("global_load_async_to_lds_b128 %0, %1, off"
                   :: "v"(dp), "v"(sp) : "memory");
    }
    for (int i = tid; i < 8 * 32; i += 256) Xs[136 * 32 + i] = 0.f;  // pad rows
    asm volatile("s_wait_asynccnt 0x0" ::: "memory");
  }
  __syncthreads();

  // stage 1: XP = X @ wrp^T + b_rp + raion_emb   (M=144,N=128,K=32)
  for (int t = w; t < 72; t += 8) {
    int mt = t >> 3, nt = t & 7;
    v8f z = {};
    v8f acc = wmma_f16(loadA_f(Xs, 32, mt * 16, 0), loadB_h(wrp_t, 128, 0, nt * 16), z);
    int n = nt * 16 + (l & 15);
#pragma unroll
    for (int i = 0; i < 8; ++i) {
      int m = mt * 16 + ((l >> 4) << 3) + i;
      int rr = (m < 136) ? m : 135;
      XP[m * 128 + n] = acc[i] + b_rp[n] + remb[rr * 128 + n];
    }
  }
  __syncthreads();

  // stage 2: QKV = XP @ w_in^T + b_in (M=144,N=384,K=128); K stored transposed
  for (int t = w; t < 216; t += 8) {
    int mt = t / 24, nt = t % 24;
    v8f acc = {};
#pragma unroll
    for (int kt = 0; kt < 4; ++kt)
      acc = wmma_f16(loadA_f(XP, 128, mt * 16, kt * 32), loadB_h(win_t, 384, kt * 32, nt * 16), acc);
    int n = nt * 16 + (l & 15);
    float bb = b_in[n];
    int mb = mt * 16 + ((l >> 4) << 3);
    if (n < 128) {
#pragma unroll
      for (int i = 0; i < 8; ++i) Qm[(size_t)(mb + i) * 128 + n] = (_Float16)(acc[i] + bb);
    } else if (n < 256) {
      v8h kv;
#pragma unroll
      for (int i = 0; i < 8; ++i) kv[i] = (_Float16)(acc[i] + bb);
      *(v8h*)(Kt + (size_t)(n - 128) * 144 + mb) = kv;   // contiguous along m
    } else {
#pragma unroll
      for (int i = 0; i < 8; ++i) Vm[(size_t)(mb + i) * 128 + (n - 256)] = (_Float16)(acc[i] + bb);
    }
  }
  __syncthreads();

  // stage 3: per-head attention, each wave owns disjoint q-tiles (rows)
  const float pw = priorw[0];
  const v16h z16 = {};
  for (int h = 0; h < 8; ++h) {
    for (int qt = w; qt < 9; qt += 8) {
      // Q fragment: head dim 16 zero-padded to K=32 (j>=8 -> k>=16 -> 0)
      v16h af;
      {
        const _Float16* qp = Qm + (size_t)(qt * 16 + (l & 15)) * 128 + h * 16 + ((l >> 4) << 3);
        v8h q8 = *(const v8h*)qp;
        v8h z8 = {};
        af = __builtin_shufflevector(q8, z8, 0, 1, 2, 3, 4, 5, 6, 7,
                                     8, 9, 10, 11, 12, 13, 14, 15);
      }
      v8f s[9];
#pragma unroll
      for (int t2 = 0; t2 < 9; ++t2) {
        // B[k][n] = K^T[h*16+k][t2*16+n]; lane&15 = n, elem s<8 -> k = (lane>>4)*8 + s, s>=8 -> pad 0
        v16h bf = z16;
#pragma unroll
        for (int s = 0; s < 8; ++s) bf[s] = Kt[(size_t)(h * 16 + ((l >> 4) << 3) + s) * 144 + t2 * 16 + (l & 15)];
        v8f z = {};
        s[t2] = wmma_f16(af, bf, z);
      }
      const int nl = l & 15;
#pragma unroll
      for (int i = 0; i < 8; ++i) {
        int m = qt * 16 + ((l >> 4) << 3) + i;
        int rq = (m < 136) ? m : 135;
        float vmax = -3.0e38f;
#pragma unroll
        for (int t2 = 0; t2 < 9; ++t2) {
          int n = t2 * 16 + nl;
          float sc = (n < 136) ? (s[t2][i] * 0.25f + pw * geo[rq * 136 + n]) : -3.0e38f;
          s[t2][i] = sc;
          vmax = fmaxf(vmax, sc);
        }
        vmax = hred_max(vmax);
        float ssum = 0.f;
#pragma unroll
        for (int t2 = 0; t2 < 9; ++t2) { float e = __expf(s[t2][i] - vmax); s[t2][i] = e; ssum += e; }
        ssum = hred_sum(ssum);
        float inv = 1.f / ssum;
#pragma unroll
        for (int t2 = 0; t2 < 9; ++t2)
          Pm[(size_t)m * 160 + t2 * 16 + nl] = (_Float16)(s[t2][i] * inv);
        Pm[(size_t)m * 160 + 144 + nl] = (_Float16)0.f;  // zero pad cols 144..159
      }
      // o = P[16,160] @ V_h[160(pad),16]
      v8f o = {};
#pragma unroll
      for (int kt = 0; kt < 5; ++kt) {
        v16h pa = loadA_h(Pm, 160, qt * 16, kt * 32);
        v16h vb;
#pragma unroll
        for (int s = 0; s < 16; ++s) {
          int kr = kt * 32 + ((s >> 3) << 4) + ((l >> 4) << 3) + (s & 7);
          vb[s] = (kr < 144) ? Vm[(size_t)kr * 128 + h * 16 + (l & 15)] : (_Float16)0.f;
        }
        o = wmma_f16(pa, vb, o);
      }
#pragma unroll
      for (int i = 0; i < 8; ++i) {
        int m = qt * 16 + ((l >> 4) << 3) + i;
        Om[(size_t)m * 128 + h * 16 + nl] = (_Float16)o[i];
      }
    }
  }
  __syncthreads();

  // stage 4: att = O @ w_o^T + b_o; nan_to_num; residual into XP
  for (int t = w; t < 72; t += 8) {
    int mt = t >> 3, nt = t & 7;
    v8f acc = {};
#pragma unroll
    for (int kt = 0; kt < 4; ++kt)
      acc = wmma_f16(loadA_h(Om, 128, mt * 16, kt * 32), loadB_h(wo_t, 128, kt * 32, nt * 16), acc);
    int n = nt * 16 + (l & 15);
#pragma unroll
    for (int i = 0; i < 8; ++i) {
      int m = mt * 16 + ((l >> 4) << 3) + i;
      float att = acc[i] + b_o[n];
      att = (att != att) ? 0.f : fminf(fmaxf(att, -3.0e38f), 3.0e38f);
      XP[m * 128 + n] += att;
    }
  }
  __syncthreads();

  // stage 5: LayerNorm rows 0..135, then mean over the 136 regions
  for (int r = w; r < 136; r += 8) {
    float vals[4]; float ssum = 0.f;
#pragma unroll
    for (int j = 0; j < 4; ++j) { vals[j] = XP[r * 128 + j * 32 + l]; ssum += vals[j]; }
    ssum = wred_sum(ssum);
    float mu = ssum * (1.f / 128.f);
    float var = 0.f;
#pragma unroll
    for (int j = 0; j < 4; ++j) { float d = vals[j] - mu; var += d * d; }
    var = wred_sum(var) * (1.f / 128.f);
    float rstd = rsqrtf(var + 1e-5f);
#pragma unroll
    for (int j = 0; j < 4; ++j) {
      int c = j * 32 + l;
      XP[r * 128 + c] = ln1g[c] * (vals[j] - mu) * rstd + ln1b[c];
    }
  }
  __syncthreads();
  if (tid < 128) {
    float ssum = 0.f;
    for (int r = 0; r < 136; ++r) ssum += XP[r * 128 + tid];
    float mval = ssum * (1.f / 136.f);
    x_agg[(size_t)bs * 128 + tid] = mval;
    x_agg_h[(size_t)bs * 128 + tid] = (_Float16)mval;
  }
}

// ---------------------------------------------------------------------------
// Generic WMMA GEMM: Y = A(f16,[Mt*16 x K]) @ Bt(f16,[K x N]) + bias
// one wave per 16x16 tile; f32 or f16 output
// ---------------------------------------------------------------------------
__global__ void __launch_bounds__(256) gemm_h_f32o(const _Float16* __restrict__ A,
                                                   const _Float16* __restrict__ Bt,
                                                   const float* __restrict__ bias,
                                                   float* __restrict__ Y,
                                                   int M, int N, int K, int Nt, int tiles) {
  int w = threadIdx.x >> 5, l = threadIdx.x & 31;
  int t = blockIdx.x * 8 + w;
  if (t >= tiles) return;
  int mt = t / Nt, nt = t % Nt;
  v8f acc = {};
  for (int k = 0; k < K; k += 32)
    acc = wmma_f16(loadA_h(A, K, mt * 16, k), loadB_h(Bt, N, k, nt * 16), acc);
  int n = nt * 16 + (l & 15);
  float bb = bias[n];
#pragma unroll
  for (int i = 0; i < 8; ++i) {
    int m = mt * 16 + ((l >> 4) << 3) + i;
    if (m < M) Y[(size_t)m * N + n] = acc[i] + bb;
  }
}

__global__ void __launch_bounds__(256) gemm_h_f16o(const _Float16* __restrict__ A,
                                                   const _Float16* __restrict__ Bt,
                                                   const float* __restrict__ bias,
                                                   _Float16* __restrict__ Y,
                                                   int M, int N, int K, int Nt, int tiles) {
  int w = threadIdx.x >> 5, l = threadIdx.x & 31;
  int t = blockIdx.x * 8 + w;
  if (t >= tiles) return;
  int mt = t / Nt, nt = t % Nt;
  v8f acc = {};
  for (int k = 0; k < K; k += 32)
    acc = wmma_f16(loadA_h(A, K, mt * 16, k), loadB_h(Bt, N, k, nt * 16), acc);
  int n = nt * 16 + (l & 15);
  float bb = bias[n];
#pragma unroll
  for (int i = 0; i < 8; ++i) {
    int m = mt * 16 + ((l >> 4) << 3) + i;
    if (m < M) Y[(size_t)m * N + n] = (_Float16)(acc[i] + bb);
  }
}

// ---------------------------------------------------------------------------
// K2b: temporal attention. grid = B(2) * heads(4); hd=32 -> K=32 = 1 WMMA step
// qkv: f16 [736][384] rows b*365+s; output pre-out-proj concat heads (f16)
// ---------------------------------------------------------------------------
__global__ void __launch_bounds__(256) temporal_attn(const _Float16* __restrict__ qkv,
                                                     _Float16* __restrict__ tatt) {
  const int b = blockIdx.x >> 2, h = blockIdx.x & 3;
  const int tid = threadIdx.x, w = tid >> 5, l = tid & 31;
  extern __shared__ char smem[];
  _Float16* Kh_t = (_Float16*)smem;          // [32][384]  K transposed
  _Float16* Vh = Kh_t + 32 * 384;            // [384][32]
  float* Pw = (float*)(Vh + 384 * 32);       // [8 waves][16][384]

  for (int i = tid; i < 384 * 32; i += 256) {
    int t = i >> 5, c = i & 31;
    _Float16 kv = (_Float16)0.f, vv = (_Float16)0.f;
    if (t < 365) {
      const _Float16* row = qkv + (size_t)(b * 365 + t) * 384;
      kv = row[128 + h * 32 + c];
      vv = row[256 + h * 32 + c];
    }
    Kh_t[(size_t)c * 384 + t] = kv;
    Vh[i] = vv;
  }
  __syncthreads();

  float* Ps = Pw + (size_t)w * 16 * 384;
  const float scale = 0.17677669529663687f;  // 1/sqrt(32)
  const v16h z16 = {};
  for (int qt = w; qt < 23; qt += 8) {
    v16h af;  // Q fragment straight from global (hd=32 = full K)
    {
      int m = qt * 16 + (l & 15);
      int rr = (m < 365) ? m : 364;
      const _Float16* qp = qkv + (size_t)(b * 365 + rr) * 384 + h * 32 + ((l >> 4) << 3);
      v8h lo = *(const v8h*)qp;
      v8h hi = *(const v8h*)(qp + 16);
      af = __builtin_shufflevector(lo, hi, 0, 1, 2, 3, 4, 5, 6, 7,
                                   8, 9, 10, 11, 12, 13, 14, 15);
      if (m >= 365) af = z16;
    }
    for (int kt = 0; kt < 23; ++kt) {
      // B[k][n] = K^T[k][kt*16+n]; lane&15 = n, elem s -> k in table order
      v16h bf;
#pragma unroll
      for (int s = 0; s < 16; ++s) bf[s] = Kh_t[(size_t)(((s >> 3) << 4) + ((l >> 4) << 3) + (s & 7)) * 384 + kt * 16 + (l & 15)];
      v8f z = {};
      v8f sc = wmma_f16(af, bf, z);
#pragma unroll
      for (int i = 0; i < 8; ++i)
        Ps[(((l >> 4) << 3) + i) * 384 + kt * 16 + (l & 15)] = sc[i] * scale;
    }
    // softmax per row (cols 0..364 valid; 365..383 -> 0 prob)
    for (int r = 0; r < 16; ++r) {
      float vmax = -3.0e38f;
      for (int c = l; c < 384; c += 32) {
        float sc = (c < 365) ? Ps[r * 384 + c] : -3.0e38f;
        vmax = fmaxf(vmax, sc);
      }
#pragma unroll
      for (int d = 1; d < 32; d <<= 1) vmax = fmaxf(vmax, __shfl_xor(vmax, d, 32));
      float ssum = 0.f;
      for (int c = l; c < 384; c += 32) {
        float sc = (c < 365) ? Ps[r * 384 + c] : -3.0e38f;
        float e = __expf(sc - vmax);
        Ps[r * 384 + c] = e;
        ssum += e;
      }
      ssum = wred_sum(ssum);
      float inv = 1.f / ssum;
      for (int c = l; c < 384; c += 32) Ps[r * 384 + c] *= inv;
    }
    // o = P[16,384] @ Vh[384,32]
    for (int ntile = 0; ntile < 2; ++ntile) {
      v8f o = {};
      for (int kt = 0; kt < 12; ++kt) {
        v16h pa = loadA_f(Ps, 384, 0, kt * 32);
        v16h vb;
#pragma unroll
        for (int s = 0; s < 16; ++s) vb[s] = Vh[(size_t)(kt * 32 + ((s >> 3) << 4) + ((l >> 4) << 3) + (s & 7)) * 32 + ntile * 16 + (l & 15)];
        o = wmma_f16(pa, vb, o);
      }
#pragma unroll
      for (int i = 0; i < 8; ++i) {
        int m = qt * 16 + ((l >> 4) << 3) + i;
        if (m < 365)
          tatt[(size_t)(b * 365 + m) * 128 + h * 32 + ntile * 16 + (l & 15)] = (_Float16)o[i];
      }
    }
  }
}

// ---------------------------------------------------------------------------
// row-wise residual+LN (f16 out) and LN+gelu (f32 out), width 128, 1 wave/row
// ---------------------------------------------------------------------------
__global__ void __launch_bounds__(256) add_ln128_h(const float* __restrict__ X,
                                                   const float* __restrict__ Yr,
                                                   const float* __restrict__ g,
                                                   const float* __restrict__ b,
                                                   _Float16* __restrict__ Oh, int M) {
  int w = threadIdx.x >> 5, l = threadIdx.x & 31;
  int r = blockIdx.x * 8 + w;
  if (r >= M) return;
  float vals[4]; float s = 0.f;
#pragma unroll
  for (int j = 0; j < 4; ++j) { int c = j * 32 + l; vals[j] = X[(size_t)r * 128 + c] + Yr[(size_t)r * 128 + c]; s += vals[j]; }
  s = wred_sum(s);
  float mu = s * (1.f / 128.f);
  float var = 0.f;
#pragma unroll
  for (int j = 0; j < 4; ++j) { float d = vals[j] - mu; var += d * d; }
  var = wred_sum(var) * (1.f / 128.f);
  float rstd = rsqrtf(var + 1e-5f);
#pragma unroll
  for (int j = 0; j < 4; ++j) { int c = j * 32 + l; Oh[(size_t)r * 128 + c] = (_Float16)(g[c] * (vals[j] - mu) * rstd + b[c]); }
}

__global__ void __launch_bounds__(256) ln_gelu128(const float* __restrict__ X,
                                                  const float* __restrict__ g,
                                                  const float* __restrict__ b,
                                                  float* __restrict__ Of, int M) {
  int w = threadIdx.x >> 5, l = threadIdx.x & 31;
  int r = blockIdx.x * 8 + w;
  if (r >= M) return;
  float vals[4]; float s = 0.f;
#pragma unroll
  for (int j = 0; j < 4; ++j) { int c = j * 32 + l; vals[j] = X[(size_t)r * 128 + c]; s += vals[j]; }
  s = wred_sum(s);
  float mu = s * (1.f / 128.f);
  float var = 0.f;
#pragma unroll
  for (int j = 0; j < 4; ++j) { float d = vals[j] - mu; var += d * d; }
  var = wred_sum(var) * (1.f / 128.f);
  float rstd = rsqrtf(var + 1e-5f);
#pragma unroll
  for (int j = 0; j < 4; ++j) { int c = j * 32 + l; Of[(size_t)r * 128 + c] = gelu_f(g[c] * (vals[j] - mu) * rstd + b[c]); }
}

// ---------------------------------------------------------------------------
// K3: cross-source fusion transformer + output head. 8 timesteps per WG ->
// 16 token rows so all matmuls are WMMA shaped; the 2x2 attention is VALU.
// ---------------------------------------------------------------------------
struct FusionW {
  const _Float16* win[2]; const float* bin[2];
  const _Float16* wo[2];  const float* bo[2];
  const float* l1g[2]; const float* l1b[2];
  const _Float16* w1[2];  const float* b1[2];
  const _Float16* w2[2];  const float* b2[2];
  const float* l2g[2]; const float* l2b[2];
  const _Float16* wo1; const float* bo1; const float* lno1g; const float* lno1b;
  const _Float16* wo2; const float* bo2; const float* lno2g; const float* lno2b;
};

__global__ void __launch_bounds__(256) fusion_kernel(
    const float* __restrict__ x_eq, const float* __restrict__ venc,
    const _Float16* __restrict__ weq_t, const float* __restrict__ b_eq,
    const float* __restrict__ type_emb, FusionW W, float* __restrict__ outp) {
  const int row0 = blockIdx.x * 8;
  const int tid = threadIdx.x, w = tid >> 5, l = tid & 31;
  extern __shared__ char smem[];
  float* T  = (float*)smem;                 // [16][128]  tokens
  float* T2 = (float*)(smem + 16 * 128 * 4);// [16][512]  scratch
  _Float16* Xeq = (_Float16*)T2;            // [16][64] alias (pre-layer only)

  for (int i = tid; i < 16 * 64; i += 256) {
    int r = i >> 6, c = i & 63;
    int ts = row0 + r;
    Xeq[i] = (r < 8 && ts < 730) ? (_Float16)x_eq[(size_t)ts * 64 + c] : (_Float16)0.f;
  }
  for (int i = tid; i < 8 * 128; i += 256) {
    int r = i >> 7, c = i & 127;
    int ts = row0 + r;
    float vv = (ts < 730) ? venc[(size_t)ts * 128 + c] : 0.f;
    T[(2 * r + 1) * 128 + c] = vv + type_emb[128 + c];
  }
  __syncthreads();
  {  // eq_enc = Xeq[16,64] @ weq^T + b_eq + type0 -> even token rows
    v8f acc = {};
    acc = wmma_f16(loadA_h(Xeq, 64, 0, 0),  loadB_h(weq_t, 128, 0,  w * 16), acc);
    acc = wmma_f16(loadA_h(Xeq, 64, 0, 32), loadB_h(weq_t, 128, 32, w * 16), acc);
    int n = w * 16 + (l & 15);
#pragma unroll
    for (int i = 0; i < 8; ++i) {
      int m = ((l >> 4) << 3) + i;
      if (m < 8) T[(2 * m) * 128 + n] = acc[i] + b_eq[n] + type_emb[n];
    }
  }
  __syncthreads();

  for (int li = 0; li < 2; ++li) {
    // qkv = T[16,128] @ w_in^T + b_in -> T2[:,0:384]
    for (int nt = w * 3; nt < w * 3 + 3; ++nt) {
      v8f acc = {};
#pragma unroll
      for (int kt = 0; kt < 4; ++kt)
        acc = wmma_f16(loadA_f(T, 128, 0, kt * 32), loadB_h(W.win[li], 384, kt * 32, nt * 16), acc);
      int n = nt * 16 + (l & 15);
#pragma unroll
      for (int i = 0; i < 8; ++i) T2[(size_t)(((l >> 4) << 3) + i) * 512 + n] = acc[i] + W.bin[li][n];
    }
    __syncthreads();
    // 2-token attention per (ts, head, query): 128 threads, hd=16, scale=0.25
    if (tid < 128) {
      int ts = tid >> 4, hh = (tid >> 1) & 7, qi = tid & 1;
      const float* q  = &T2[(2 * ts + qi) * 512 + hh * 16];
      const float* k0 = &T2[(2 * ts + 0) * 512 + 128 + hh * 16];
      const float* k1 = &T2[(2 * ts + 1) * 512 + 128 + hh * 16];
      float s0 = 0.f, s1 = 0.f;
#pragma unroll
      for (int c = 0; c < 16; ++c) { s0 += q[c] * k0[c]; s1 += q[c] * k1[c]; }
      s0 *= 0.25f; s1 *= 0.25f;
      float mx = fmaxf(s0, s1);
      float e0 = __expf(s0 - mx), e1 = __expf(s1 - mx);
      float inv = 1.f / (e0 + e1);
      e0 *= inv; e1 *= inv;
      const float* v0 = &T2[(2 * ts + 0) * 512 + 256 + hh * 16];
      const float* v1 = &T2[(2 * ts + 1) * 512 + 256 + hh * 16];
      float* op = &T2[(2 * ts + qi) * 512 + 384 + hh * 16];
#pragma unroll
      for (int c = 0; c < 16; ++c) op[c] = e0 * v0[c] + e1 * v1[c];
    }
    __syncthreads();
    {  // out-proj + residual
      v8f acc = {};
#pragma unroll
      for (int kt = 0; kt < 4; ++kt)
        acc = wmma_f16(loadA_f(T2 + 384, 512, 0, kt * 32), loadB_h(W.wo[li], 128, kt * 32, w * 16), acc);
      int n = w * 16 + (l & 15);
#pragma unroll
      for (int i = 0; i < 8; ++i) { int m = ((l >> 4) << 3) + i; T[m * 128 + n] += acc[i] + W.bo[li][n]; }
    }
    __syncthreads();
    for (int r = w * 2; r < w * 2 + 2; ++r) wave_ln_lds(&T[r * 128], 128, W.l1g[li], W.l1b[li], false);
    __syncthreads();
    // FFN1: gelu(T @ w1^T + b1) -> T2[16,512]
    for (int nt = w * 4; nt < w * 4 + 4; ++nt) {
      v8f acc = {};
#pragma unroll
      for (int kt = 0; kt < 4; ++kt)
        acc = wmma_f16(loadA_f(T, 128, 0, kt * 32), loadB_h(W.w1[li], 512, kt * 32, nt * 16), acc);
      int n = nt * 16 + (l & 15);
#pragma unroll
      for (int i = 0; i < 8; ++i) T2[(size_t)(((l >> 4) << 3) + i) * 512 + n] = gelu_f(acc[i] + W.b1[li][n]);
    }
    __syncthreads();
    {  // FFN2 + residual
      v8f acc = {};
#pragma unroll
      for (int kt = 0; kt < 16; ++kt)
        acc = wmma_f16(loadA_f(T2, 512, 0, kt * 32), loadB_h(W.w2[li], 128, kt * 32, w * 16), acc);
      int n = w * 16 + (l & 15);
#pragma unroll
      for (int i = 0; i < 8; ++i) { int m = ((l >> 4) << 3) + i; T[m * 128 + n] += acc[i] + W.b2[li][n]; }
    }
    __syncthreads();
    for (int r = w * 2; r < w * 2 + 2; ++r) wave_ln_lds(&T[r * 128], 128, W.l2g[li], W.l2b[li], false);
    __syncthreads();
  }

  // output head: cat [8(ts) x 256] (padded to 16 rows)
  float* Tc = T2;             // [16][256]
  float* H1 = T2 + 16 * 256;  // [16][256]
  for (int i = tid; i < 16 * 256; i += 256) {
    int r = i >> 8, c = i & 255;
    Tc[i] = (r < 8) ? T[(2 * r + (c >> 7)) * 128 + (c & 127)] : 0.f;
  }
  __syncthreads();
  for (int nt = w * 2; nt < w * 2 + 2; ++nt) {  // h1 = cat @ w_o1^T + b_o1
    v8f acc = {};
#pragma unroll
    for (int kt = 0; kt < 8; ++kt)
      acc = wmma_f16(loadA_f(Tc, 256, 0, kt * 32), loadB_h(W.wo1, 256, kt * 32, nt * 16), acc);
    int n = nt * 16 + (l & 15);
#pragma unroll
    for (int i = 0; i < 8; ++i) H1[(((l >> 4) << 3) + i) * 256 + n] = acc[i] + W.bo1[n];
  }
  __syncthreads();
  wave_ln_lds(&H1[w * 256], 256, W.lno1g, W.lno1b, true);  // LN_o1 + gelu, rows 0..7
  __syncthreads();
  {  // out = h1 @ w_o2^T + b_o2 -> Tc[16,128]
    v8f acc = {};
#pragma unroll
    for (int kt = 0; kt < 8; ++kt)
      acc = wmma_f16(loadA_f(H1, 256, 0, kt * 32), loadB_h(W.wo2, 128, kt * 32, w * 16), acc);
    int n = w * 16 + (l & 15);
#pragma unroll
    for (int i = 0; i < 8; ++i) Tc[(((l >> 4) << 3) + i) * 128 + n] = acc[i] + W.bo2[n];
  }
  __syncthreads();
  {  // LN_o2 + store (wave w owns timestep row0+w)
    wave_ln_lds(&Tc[w * 128], 128, W.lno2g, W.lno2b, false);
    int ts = row0 + w;
    if (ts < 730)
      for (int c = l; c < 128; c += 32) outp[(size_t)ts * 128 + c] = Tc[w * 128 + c];
  }
}

// ---------------------------------------------------------------------------
// host
// ---------------------------------------------------------------------------
extern "C" void kernel_launch(void* const* d_in, const int* in_sizes, int n_in,
                              void* d_out, int out_size, void* d_ws, size_t ws_size,
                              hipStream_t stream) {
  (void)in_sizes; (void)n_in; (void)out_size; (void)ws_size;
  char* ws = (char*)d_ws;
  size_t off = 0;
  auto alloc = [&](size_t bytes) { char* p = ws + off; off += (bytes + 255) & ~(size_t)255; return p; };

  // f16 transposed weights (Wt[in][out])
  _Float16* wrp_t  = (_Float16*)alloc(32 * 128 * 2);
  _Float16* wcin_t = (_Float16*)alloc(128 * 384 * 2);
  _Float16* wco_t  = (_Float16*)alloc(128 * 128 * 2);
  _Float16* wtin_t = (_Float16*)alloc(128 * 384 * 2);
  _Float16* wto_t  = (_Float16*)alloc(128 * 128 * 2);
  _Float16* wout_t = (_Float16*)alloc(128 * 128 * 2);
  _Float16* weq_t  = (_Float16*)alloc(64 * 128 * 2);
  _Float16* wlin[2], *wlo[2], *wl1[2], *wl2[2];
  for (int li = 0; li < 2; ++li) {
    wlin[li] = (_Float16*)alloc(128 * 384 * 2);
    wlo[li]  = (_Float16*)alloc(128 * 128 * 2);
    wl1[li]  = (_Float16*)alloc(128 * 512 * 2);
    wl2[li]  = (_Float16*)alloc(512 * 128 * 2);
  }
  _Float16* wo1_t = (_Float16*)alloc(256 * 256 * 2);
  _Float16* wo2_t = (_Float16*)alloc(256 * 128 * 2);
  // activations (730 rows padded to 736)
  float*    x_agg   = (float*)alloc(736 * 128 * 4);
  _Float16* x_agg_h = (_Float16*)alloc(736 * 128 * 2);
  _Float16* qkv_h   = (_Float16*)alloc(736 * 384 * 2);
  _Float16* tatt_h  = (_Float16*)alloc(736 * 128 * 2);
  float*    tproj   = (float*)alloc(736 * 128 * 4);
  _Float16* xl_h    = (_Float16*)alloc(736 * 128 * 2);
  float*    hbuf    = (float*)alloc(736 * 128 * 4);
  float*    venc    = (float*)alloc(736 * 128 * 4);

  auto XPOSE = [&](int idx, int rows, int cols, _Float16* dst) {
    int total = rows * cols;
    cvt_transpose<<<dim3((total + 255) / 256), dim3(256), 0, stream>>>(
        (const float*)d_in[idx], dst, rows, cols);
  };
  // d_in order: x_viina, x_equipment, geo_prior, then enc dict, then fus dict (DFS)
  XPOSE(3, 128, 32, wrp_t);
  XPOSE(7, 384, 128, wcin_t);   XPOSE(9, 128, 128, wco_t);
  XPOSE(11, 384, 128, wtin_t);  XPOSE(13, 128, 128, wto_t);
  XPOSE(19, 128, 128, wout_t);
  XPOSE(23, 128, 64, weq_t);
  XPOSE(26, 384, 128, wlin[0]); XPOSE(28, 128, 128, wlo[0]);
  XPOSE(32, 512, 128, wl1[0]);  XPOSE(34, 128, 512, wl2[0]);
  XPOSE(38, 384, 128, wlin[1]); XPOSE(40, 128, 128, wlo[1]);
  XPOSE(44, 512, 128, wl1[1]);  XPOSE(46, 128, 512, wl2[1]);
  XPOSE(50, 256, 256, wo1_t);   XPOSE(54, 128, 256, wo2_t);

  // K1: geographic encoder (one WG per (b,s)); 267264 B dynamic LDS
  encode_regions<<<dim3(730), dim3(256), 267264, stream>>>(
      (const float*)d_in[0], (const float*)d_in[2], (const float*)d_in[6],
      wrp_t, (const float*)d_in[4], (const float*)d_in[5],
      wcin_t, (const float*)d_in[8], wco_t, (const float*)d_in[10],
      (const float*)d_in[15], (const float*)d_in[16], x_agg, x_agg_h);

  // K2: temporal attention over S=365 per batch
  gemm_h_f16o<<<dim3((46 * 24 + 7) / 8), dim3(256), 0, stream>>>(
      x_agg_h, wtin_t, (const float*)d_in[12], qkv_h, 730, 384, 128, 24, 46 * 24);
  temporal_attn<<<dim3(8), dim3(256), 245760, stream>>>(qkv_h, tatt_h);
  gemm_h_f32o<<<dim3((46 * 8 + 7) / 8), dim3(256), 0, stream>>>(
      tatt_h, wto_t, (const float*)d_in[14], tproj, 730, 128, 128, 8, 46 * 8);
  add_ln128_h<<<dim3(92), dim3(256), 0, stream>>>(
      x_agg, tproj, (const float*)d_in[17], (const float*)d_in[18], xl_h, 730);
  gemm_h_f32o<<<dim3((46 * 8 + 7) / 8), dim3(256), 0, stream>>>(
      xl_h, wout_t, (const float*)d_in[20], hbuf, 730, 128, 128, 8, 46 * 8);
  ln_gelu128<<<dim3(92), dim3(256), 0, stream>>>(
      hbuf, (const float*)d_in[21], (const float*)d_in[22], venc, 730);

  // K3: fusion transformer + output head
  FusionW W;
  W.win[0] = wlin[0]; W.bin[0] = (const float*)d_in[27];
  W.wo[0]  = wlo[0];  W.bo[0]  = (const float*)d_in[29];
  W.l1g[0] = (const float*)d_in[30]; W.l1b[0] = (const float*)d_in[31];
  W.w1[0]  = wl1[0];  W.b1[0]  = (const float*)d_in[33];
  W.w2[0]  = wl2[0];  W.b2[0]  = (const float*)d_in[35];
  W.l2g[0] = (const float*)d_in[36]; W.l2b[0] = (const float*)d_in[37];
  W.win[1] = wlin[1]; W.bin[1] = (const float*)d_in[39];
  W.wo[1]  = wlo[1];  W.bo[1]  = (const float*)d_in[41];
  W.l1g[1] = (const float*)d_in[42]; W.l1b[1] = (const float*)d_in[43];
  W.w1[1]  = wl1[1];  W.b1[1]  = (const float*)d_in[45];
  W.w2[1]  = wl2[1];  W.b2[1]  = (const float*)d_in[47];
  W.l2g[1] = (const float*)d_in[48]; W.l2b[1] = (const float*)d_in[49];
  W.wo1 = wo1_t; W.bo1 = (const float*)d_in[51];
  W.lno1g = (const float*)d_in[52]; W.lno1b = (const float*)d_in[53];
  W.wo2 = wo2_t; W.bo2 = (const float*)d_in[55];
  W.lno2g = (const float*)d_in[56]; W.lno2b = (const float*)d_in[57];

  fusion_kernel<<<dim3(92), dim3(256), 40960, stream>>>(
      (const float*)d_in[1], venc, weq_t, (const float*)d_in[24],
      (const float*)d_in[25], W, (float*)d_out);
}
